// RoutingCapsules_23416161698533
// MI455X (gfx1250) — matrix operand order repaired and checked by reading, not compiled, not run on hardware
//
#include <hip/hip_runtime.h>

typedef __attribute__((ext_vector_type(16))) _Float16 v16h;
typedef __attribute__((ext_vector_type(8)))  _Float16 v8h;
typedef __attribute__((ext_vector_type(2)))  _Float16 v2h;
typedef __attribute__((ext_vector_type(8)))  float    v8f;

#define CB 64     // batch
#define CC 32     // classes
#define CN 2048   // input caps
#define CD 32     // out dim
#define CI 16     // in dim

// workspace layout (bytes)
#define OFF_WH   ((size_t)0)                       // f16 W, d-major [C][N][D][I]  : 67108864
#define OFF_WHT  ((size_t)67108864)                // f16 W, i-major [C][N][I][D]  : 67108864
#define OFF_XH   ((size_t)134217728)               // f16 x        [B][N][I]      : 4194304
#define OFF_S    ((size_t)138412032)               // f32 s accum  [B][C][D]      : 262144
#define OFF_VA   ((size_t)138674176)               // f32 v_A      [B][C][D]      : 262144
#define OFF_VSH  ((size_t)138936320)               // f16 vsum     [B][C][D]      : 131072

// -------- conversion: one block per (c,n) tile; coalesced in, coalesced out ----
__global__ __launch_bounds__(256) void conv_w2(const float* __restrict__ W,
                                               _Float16* __restrict__ Wh,
                                               _Float16* __restrict__ WhT) {
  __shared__ __align__(16) _Float16 tile[16 * 32];   // [i][d]
  const int t = threadIdx.x;
  const size_t cn = blockIdx.x;                      // 0 .. C*N-1
  const float* src = W + cn * (CD * CI);             // 512 f32, row-major [d][i]
  const int e = t * 2;
  const int d = e >> 4, i = e & 15;
  const float2 f = *(const float2*)(src + e);
  const _Float16 h0 = (_Float16)f.x, h1 = (_Float16)f.y;
  v2h pk; pk[0] = h0; pk[1] = h1;
  *(v2h*)(Wh + cn * (CD * CI) + e) = pk;             // d-major copy
  tile[i * 32 + d]       = h0;                       // transpose through LDS
  tile[(i + 1) * 32 + d] = h1;
  __syncthreads();
  v2h o = *(const v2h*)(tile + e);
  *(v2h*)(WhT + cn * (CD * CI) + e) = o;             // i-major copy
}

__global__ __launch_bounds__(256) void conv_x(const float* __restrict__ x,
                                              _Float16* __restrict__ Xh) {
  size_t idx = (size_t)blockIdx.x * 256 + threadIdx.x;   // over B*N rows of 16
  if (idx >= (size_t)CB * CN) return;
  const float* src = x + idx * CI;
  v8h lo, hi;
#pragma unroll
  for (int i = 0; i < 8; ++i) { lo[i] = (_Float16)src[i]; hi[i] = (_Float16)src[8 + i]; }
  *(v8h*)(Xh + idx * CI)     = lo;
  *(v8h*)(Xh + idx * CI + 8) = hi;
}

__global__ __launch_bounds__(256) void zero_s(float* __restrict__ s) {
  int t = blockIdx.x * 256 + threadIdx.x;
  if (t < CB * CC * CD) s[t] = 0.f;
}

// ---- pass 0: s0 = sum_n (W @ x) as ONE GEMM with K=(n,i); 2 n's per K=32 step ----
__global__ __launch_bounds__(256) void uhat_pass0(const _Float16* __restrict__ Wh,
                                                  const _Float16* __restrict__ Xh,
                                                  float* __restrict__ s) {
  const int tid = threadIdx.x, wave = tid >> 5, lane = tid & 31;
  const int hl = lane >> 4, r15 = lane & 15;
  const int mtile = blockIdx.x * 2 + (wave >> 2);   // 64 m-tiles over M=C*D=1024
  const int btile = wave & 3;                       // 4 b-tiles over B=64
  const int m = mtile * 16 + r15;
  const int c = m >> 5, d = m & 31;
  const int bg = btile * 16 + r15;
  const int nb = blockIdx.y * 64;

  v8f acc = {};
  for (int ns = 0; ns < 64; ns += 2) {
    const int n0 = nb + ns, n1 = n0 + 1;
    // A rows (c,d): lanes<16 a[0..15] = (n0, i=0..15), lanes>=16 a[0..15] = (n1, i=0..15)
    v16h a = *(const v16h*)(Wh + (((size_t)c * CN + (hl ? n1 : n0)) * CD + d) * CI);
    // B cols b: lanes<16 K=0..15 -> x[b,n0,:], lanes>=16 K=16..31 -> x[b,n1,:]
    v16h b = *(const v16h*)(Xh + ((size_t)bg * CN + (hl ? n1 : n0)) * CI);
    acc = __builtin_amdgcn_wmma_f32_16x16x32_f16(false, a, false, b, (short)0, acc,
                                                 false, false);
    __builtin_prefetch(Wh + (((size_t)c * CN + (n0 + 2)) * CD + d) * CI, 0, 3);
  }
#pragma unroll
  for (int rr = 0; rr < 8; ++rr) {
    const int mm = mtile * 16 + rr + 8 * hl;
    atomicAdd(&s[((size_t)bg * CC + (mm >> 5)) * CD + (mm & 31)], acc[rr]);
  }
}

// ---------------- squash / v updates ----------------
__global__ __launch_bounds__(256) void squash_k(float* __restrict__ s,
                                                float* __restrict__ vA,
                                                _Float16* __restrict__ vsumh,
                                                float* __restrict__ out, int mode) {
  int t = blockIdx.x * 256 + threadIdx.x;          // 65536 = B*C*D; 32 lanes = one (b,c) row
  float val = s[t];
  if (mode == 0) val *= (1.0f / (float)CC);        // softmax(0) is uniform 1/C
  float sq = val * val;
#pragma unroll
  for (int off = 16; off >= 1; off >>= 1) sq += __shfl_xor(sq, off, 32);
  float v = val * (sq / (1.0f + sq)) * rsqrtf(sq + 1e-8f);
  if (mode == 0)      { vA[t] = v; vsumh[t] = (_Float16)v;             s[t] = 0.f; }
  else if (mode == 1) { float vs = vA[t] + v; vsumh[t] = (_Float16)vs; s[t] = 0.f; }
  else                { out[t] = v; }
}

// ---------------- routing pass: logits via G-GEMM, s via y-GEMM ----------------
__global__ __launch_bounds__(256) void routing_pass(const _Float16* __restrict__ Wh,
                                                    const _Float16* __restrict__ WhT,
                                                    const _Float16* __restrict__ Xh,
                                                    const _Float16* __restrict__ vsumh,
                                                    float* __restrict__ sacc_g) {
  __shared__ __align__(16) _Float16 xsh[16][2][16];   // [b][nn][i], 1 KB
  __shared__ float lgt[2][CC][16];
  __shared__ float sfx[2][CC][16];

  const int btile = blockIdx.x;                    // 0..3
  const int nbase = blockIdx.y * 32;               // Tn = 32
  const int tid = threadIdx.x, wave = tid >> 5, lane = tid & 31;
  const int hl = lane >> 4, r15 = lane & 15;
  const int c0 = wave * 4;                         // each wave owns 4 classes
  const int bg = btile * 16 + r15;

  // A operands for G-GEMM: rows b, K = d (32) of vsum, per owned class
  v16h aG[4];
#pragma unroll
  for (int j = 0; j < 4; ++j) {
    const _Float16* vp = vsumh + ((size_t)bg * CC + (c0 + j)) * CD + 16 * hl;
    v8h p = *(const v8h*)vp;
    v8h q = *(const v8h*)(vp + 8);
#pragma unroll
    for (int e = 0; e < 8; ++e) { aG[j][e] = p[e]; aG[j][8 + e] = q[e]; }
  }

  v8f sc[4][2];
#pragma unroll
  for (int j = 0; j < 4; ++j)
#pragma unroll
    for (int dt = 0; dt < 2; ++dt) sc[j][dt] = {};

  for (int ns = 0; ns < 32; ns += 2) {
    const int n0 = nbase + ns, n1 = n0 + 1;

    // stage raw f16 x tile (16 b x 2 n x 16 i = 1 KB) via async copy to LDS:
    // per row b, x[b,n0,0..15] and x[b,n1,0..15] are 64 contiguous bytes.
    if (tid < 64) {
      const int bb = tid >> 2, part = tid & 3;
      unsigned long long gp = (unsigned long long)(const void*)
          (Xh + ((size_t)(btile * 16 + bb) * CN + n0) * CI + part * 8);
      unsigned lds_off = (unsigned)(uintptr_t)(&xsh[bb][0][0]) + part * 16u;
      asm volatile("global_load_async_to_lds_b128 %0, %1, off"
                   :: "v"(lds_off), "v"(gp) : "memory");
    }
    asm volatile("s_wait_asynccnt 0" ::: "memory");
    __syncthreads();

    // G = vsum @ W  (per class, per n), then logits[b,c,n] = sum_i x*G
#pragma unroll
    for (int j = 0; j < 4; ++j) {
      const int cc = c0 + j;
#pragma unroll
      for (int nn = 0; nn < 2; ++nn) {
        const int n = n0 + nn;
        v16h bop = *(const v16h*)(WhT + (((size_t)cc * CN + n) * CI + r15) * CD +
                                  hl * 16);
        v8f g = {};
        g = __builtin_amdgcn_wmma_f32_16x16x32_f16(false, aG[j], false, bop,
                                                   (short)0, g, false, false);
#pragma unroll
        for (int rr = 0; rr < 8; ++rr) {
          float t = g[rr] * (float)xsh[rr + 8 * hl][nn][r15];
          t += __shfl_xor(t, 1, 16);
          t += __shfl_xor(t, 2, 16);
          t += __shfl_xor(t, 4, 16);
          t += __shfl_xor(t, 8, 16);
          if (r15 == 0) lgt[nn][cc][rr + 8 * hl] = t;
        }
      }
      __builtin_prefetch(WhT + (((size_t)cc * CN + (n0 + 2)) * CI + r15) * CD, 0, 3);
    }
    __syncthreads();

    // softmax over 32 classes: 8 consecutive lanes cooperate per (nn,b) pair
    {
      const int pair = tid >> 3, sub = tid & 7;    // 32 pairs x 8 lanes
      const int nn = pair >> 4, bb = pair & 15;
      float l[4];
#pragma unroll
      for (int k = 0; k < 4; ++k) l[k] = lgt[nn][sub * 4 + k][bb];
      float mx = fmaxf(fmaxf(l[0], l[1]), fmaxf(l[2], l[3]));
      mx = fmaxf(mx, __shfl_xor(mx, 1, 8));
      mx = fmaxf(mx, __shfl_xor(mx, 2, 8));
      mx = fmaxf(mx, __shfl_xor(mx, 4, 8));
      float e[4], sum = 0.f;
#pragma unroll
      for (int k = 0; k < 4; ++k) { e[k] = __expf(l[k] - mx); sum += e[k]; }
      sum += __shfl_xor(sum, 1, 8);
      sum += __shfl_xor(sum, 2, 8);
      sum += __shfl_xor(sum, 4, 8);
      const float inv = 1.f / sum;
#pragma unroll
      for (int k = 0; k < 4; ++k) sfx[nn][sub * 4 + k][bb] = e[k] * inv;
    }
    __syncthreads();

    // s += y @ W with y = softmax*x, K = (n0,n1)x(i) = 32
#pragma unroll
    for (int j = 0; j < 4; ++j) {
      const int cc = c0 + j;
      _Float16 h0 = (_Float16)sfx[hl][cc][r15];
      _Float16 h1 = (_Float16)sfx[hl][cc][r15];
      v8h x0 = *(const v8h*)(&xsh[r15][hl][0]);
      v8h x1 = *(const v8h*)(&xsh[r15][hl][8]);
      v16h ay;
#pragma unroll
      for (int e = 0; e < 8; ++e) { ay[e] = x0[e] * h0; ay[8 + e] = x1[e] * h1; }
#pragma unroll
      for (int dt = 0; dt < 2; ++dt) {
        const int d = dt * 16 + r15;
        v16h bop = *(const v16h*)(Wh + (((size_t)cc * CN + (hl ? n1 : n0)) * CD + d) * CI);
        sc[j][dt] = __builtin_amdgcn_wmma_f32_16x16x32_f16(false, ay, false, bop,
                                                           (short)0, sc[j][dt],
                                                           false, false);
      }
      __builtin_prefetch(Wh + (((size_t)cc * CN + (n0 + 2)) * CD + r15) * CI, 0, 3);
    }
    __syncthreads();
  }

  // flush per-chunk s accumulators
#pragma unroll
  for (int j = 0; j < 4; ++j) {
    const int cc = c0 + j;
#pragma unroll
    for (int dt = 0; dt < 2; ++dt) {
      const int d = dt * 16 + r15;
#pragma unroll
      for (int rr = 0; rr < 8; ++rr) {
        const int bb = btile * 16 + rr + 8 * hl;
        atomicAdd(&sacc_g[((size_t)bb * CC + cc) * CD + d], sc[j][dt][rr]);
      }
    }
  }
}

extern "C" void kernel_launch(void* const* d_in, const int* in_sizes, int n_in,
                              void* d_out, int out_size, void* d_ws, size_t ws_size,
                              hipStream_t stream) {
  const float* x = (const float*)d_in[0];   // [B,N,I]
  const float* W = (const float*)d_in[1];   // [C,N,D,I]
  float* out = (float*)d_out;               // [B,C,D]
  char* ws = (char*)d_ws;
  _Float16* Wh    = (_Float16*)(ws + OFF_WH);
  _Float16* WhT   = (_Float16*)(ws + OFF_WHT);
  _Float16* Xh    = (_Float16*)(ws + OFF_XH);
  float*    s     = (float*)(ws + OFF_S);
  float*    vA    = (float*)(ws + OFF_VA);
  _Float16* vsumh = (_Float16*)(ws + OFF_VSH);

  conv_w2<<<CC * CN, 256, 0, stream>>>(W, Wh, WhT);
  conv_x<<<(CB * CN + 255) / 256, 256, 0, stream>>>(x, Xh);
  zero_s<<<(CB * CC * CD + 255) / 256, 256, 0, stream>>>(s);

  // step 0: uniform routing -> s0, v_A
  uhat_pass0<<<dim3(32, 32), 256, 0, stream>>>(Wh, Xh, s);
  squash_k<<<CB * CC * CD / 256, 256, 0, stream>>>(s, vA, vsumh, out, 0);

  // routing iteration 1: logits = du . v_A
  routing_pass<<<dim3(4, 64), 256, 0, stream>>>(Wh, WhT, Xh, vsumh, s);
  squash_k<<<CB * CC * CD / 256, 256, 0, stream>>>(s, vA, vsumh, out, 1);

  // routing iteration 2 (final): logits = du . (v_A + v_B), output = squash(s)
  routing_pass<<<dim3(4, 64), 256, 0, stream>>>(Wh, WhT, Xh, vsumh, s);
  squash_k<<<CB * CC * CD / 256, 256, 0, stream>>>(s, vA, vsumh, out, 2);
}
